// CIN_87823491268645
// MI455X (gfx1250) — compile-verified, matrix operand order checked by reading, not run on hardware
//
#include <hip/hip_runtime.h>
#include <math.h>

typedef __bf16 v16bf __attribute__((ext_vector_type(16)));
typedef __bf16 v8bf  __attribute__((ext_vector_type(8)));
typedef float  v8f   __attribute__((ext_vector_type(8)));
typedef float  v4f   __attribute__((ext_vector_type(4)));

namespace {
constexpr int kF0     = 40;
constexpr int kF      = 40;
constexpr int kD      = 64;
constexpr int kM      = 128;
constexpr int kK      = kF0 * kF;       // 1600
constexpr int kKC     = 32;             // K per v_wmma_f32_16x16x32_bf16
constexpr int kNChunk = kK / kKC;       // 50
constexpr int kNT     = kM / 16;        // 8 N-tiles
constexpr int kNB     = 4;              // b values per workgroup
constexpr int kThreads = 256;           // 8 waves
constexpr int kXElems  = kF0 * kD;      // 2560 floats per (b, array)
constexpr int kChunkElems = kNT * 32 * 16;          // 4096 bf16 = 8 KB
constexpr int kChunkBytes = kChunkElems * 2;
constexpr int kPackElems  = kNChunk * kChunkElems;  // 204800 (409,600 B)
// dynamic LDS layout (bytes)
constexpr int kX0Off   = 0;                          // 4*2560 f32 = 40960
constexpr int kXsOff   = kX0Off + kNB * kXElems * 4; // 40960
constexpr int kKbufOff = kXsOff + kNB * kXElems * 4; // 81920, 2*8KB bf16
constexpr int kLdsBytes = kKbufOff + 2 * kChunkBytes; // 98304 (96 KB)
// obuf (32 KB, f32[8192]) aliases offset 0 after compute completes
}

// ---------------------------------------------------------------------------
// Pre-pack: K[1600,128] f32 -> bf16 in WMMA-B fragment order (ISA 7.12.2).
// (c, n, lane l, half h): value = K[c*32 + kk(h,l), n*16 + (l&15)],
// kk = ((h<8)?h:h+8) + (l>=16 ? 8 : 0). Lane's 16 halves = 32 contiguous B.
// ---------------------------------------------------------------------------
__global__ __launch_bounds__(256)
void prepack_kernel_b(const float* __restrict__ kg, __bf16* __restrict__ ws)
{
  const int tid = blockIdx.x * 256 + threadIdx.x;
  if (tid >= kPackElems) return;
  const int h = tid & 15;
  const int l = (tid >> 4) & 31;
  const int n = (tid >> 9) & 7;
  const int c = tid >> 12;
  const int kk  = ((h < 8) ? h : (h + 8)) + ((l >> 4) << 3);
  ws[tid] = (__bf16)kg[(c * kKC + kk) * kM + (n * 16 + (l & 15))];
}

// ---------------------------------------------------------------------------
// CDNA5 async copy: global -> LDS, 16 bytes, GVS addressing, ASYNCcnt-tracked.
// ---------------------------------------------------------------------------
__device__ __forceinline__ void async_g2l_b128(uint32_t lds_off,
                                               const void* sbase,
                                               uint32_t voff)
{
  asm volatile("global_load_async_to_lds_b128 %0, %1, %2 offset:0"
               :: "v"(lds_off), "v"(voff), "s"(sbase) : "memory");
}
__device__ __forceinline__ void wait_asynccnt0()
{
  asm volatile("s_wait_asynccnt 0" ::: "memory");
}

union BTile { v16bf v; v8bf h[2]; };

// A fragment: O[d, ij] = x0[b,i,d]*x[b,j,d] in bf16 (ISA 7.12.2 A layout).
__device__ __forceinline__ v16bf build_afrag(const float* __restrict__ x0p,
                                             const float* __restrict__ xp,
                                             int c, int kOff, int dA)
{
  v16bf a;
#pragma unroll
  for (int h = 0; h < 16; ++h) {
    const int kk = ((h < 8) ? h : (h + 8)) + kOff;
    const int ij = c * kKC + kk;
    const int i  = ij / kF;
    const int j  = ij - i * kF;
    a[h] = (__bf16)(x0p[i * kD + dA] * xp[j * kD + dA]);
  }
  return a;
}

__global__ __launch_bounds__(kThreads)
void cin_wmma_bf16_kernel(const float* __restrict__ x0g,
                          const float* __restrict__ xg,
                          const __bf16* __restrict__ bpack,
                          float* __restrict__ outg)
{
  extern __shared__ char smraw[];
  float*  x0s  = reinterpret_cast<float*>(smraw + kX0Off);   // [4][40][64]
  float*  xs   = reinterpret_cast<float*>(smraw + kXsOff);   // [4][40][64]
  __bf16* kbuf = reinterpret_cast<__bf16*>(smraw + kKbufOff);// [2][4096]
  float*  obuf = reinterpret_cast<float*>(smraw);            // alias, epilogue

  const int t       = threadIdx.x;
  const int wg      = blockIdx.x;
  const int lane    = t & 31;
  const int wave    = t >> 5;          // 0..7
  const int halfSel = lane >> 4;
  const int l16     = lane & 15;
  const int kOff    = halfSel * 8;
  const int bLocal  = wave >> 1;       // 0..3 : which b this wave works on
  const int dHalf   = (wave & 1) * 32; // two 16-row tiles at dHalf, dHalf+16
  const float* x0p  = x0s + bLocal * kXElems;
  const float* xp   = xs  + bLocal * kXElems;

  // ---- async-stage B chunk 0 into LDS buffer 0 (2 x b128 per thread) ----
  {
    const uint32_t ldsB = (uint32_t)(uintptr_t)kbuf;   // AS3 ptr == LDS offset
    const char* g = (const char*)bpack;                // chunk 0 base
    async_g2l_b128(ldsB + t * 16,        g, t * 16);
    async_g2l_b128(ldsB + t * 16 + 4096, g, t * 16 + 4096);
  }

  // ---- stage x0/x for 4 consecutive b (contiguous 10240 floats each) ----
  {
    const v4f* x0g4 = reinterpret_cast<const v4f*>(x0g + (size_t)wg * (kNB * kXElems));
    const v4f* xg4  = reinterpret_cast<const v4f*>(xg  + (size_t)wg * (kNB * kXElems));
    v4f* x0s4 = reinterpret_cast<v4f*>(x0s);
    v4f* xs4  = reinterpret_cast<v4f*>(xs);
#pragma unroll
    for (int r = 0; r < 10; ++r) {          // 2560 float4 per array
      const int q = t + kThreads * r;
      x0s4[q] = x0g4[q];
      xs4[q]  = xg4[q];
    }
  }

  v8f acc[2][kNT];
#pragma unroll
  for (int rt = 0; rt < 2; ++rt)
#pragma unroll
    for (int n = 0; n < kNT; ++n)
      acc[rt][n] = (v8f){0.f, 0.f, 0.f, 0.f, 0.f, 0.f, 0.f, 0.f};

  // ---- main loop: 1 barrier + 1 async-wait per chunk, branch-free body ----
  const uint32_t kbufLds = (uint32_t)(uintptr_t)kbuf;
  for (int c = 0; c < kNChunk - 1; ++c) {
    wait_asynccnt0();                  // my chunk-c async writes landed
    __syncthreads();                   // all waves' writes visible; prev reads done
    {                                  // async-prefetch chunk c+1 -> other buffer
      const uint32_t dst = kbufLds + ((c + 1) & 1) * kChunkBytes;
      const char* g = (const char*)bpack + (size_t)(c + 1) * kChunkBytes;
      async_g2l_b128(dst + t * 16,        g, t * 16);
      async_g2l_b128(dst + t * 16 + 4096, g, t * 16 + 4096);
    }
    const __bf16* kb = kbuf + (c & 1) * kChunkElems;
    const v16bf a0 = build_afrag(x0p, xp, c, kOff, dHalf + l16);
    const v16bf a1 = build_afrag(x0p, xp, c, kOff, dHalf + 16 + l16);
#pragma unroll
    for (int gtile = 0; gtile < 2; ++gtile) {      // 2 groups of 4 tiles
      BTile bt[4];
#pragma unroll
      for (int n = 0; n < 4; ++n) {                // 8 ds_load_b128 clause
        const __bf16* bp = kb + ((gtile * 4 + n) * 32 + lane) * 16;
        bt[n].h[0] = *reinterpret_cast<const v8bf*>(bp);
        bt[n].h[1] = *reinterpret_cast<const v8bf*>(bp + 8);
      }
#pragma unroll
      for (int n = 0; n < 4; ++n) {                // 8 back-to-back WMMAs
        acc[0][gtile * 4 + n] = __builtin_amdgcn_wmma_f32_16x16x32_bf16(
            false, a0, false, bt[n].v, (short)0, acc[0][gtile * 4 + n], false, false);
        acc[1][gtile * 4 + n] = __builtin_amdgcn_wmma_f32_16x16x32_bf16(
            false, a1, false, bt[n].v, (short)0, acc[1][gtile * 4 + n], false, false);
      }
    }
  }
  {                                     // peeled last chunk (no prefetch)
    const int c = kNChunk - 1;
    wait_asynccnt0();
    __syncthreads();
    const __bf16* kb = kbuf + (c & 1) * kChunkElems;
    const v16bf a0 = build_afrag(x0p, xp, c, kOff, dHalf + l16);
    const v16bf a1 = build_afrag(x0p, xp, c, kOff, dHalf + 16 + l16);
#pragma unroll
    for (int gtile = 0; gtile < 2; ++gtile) {
      BTile bt[4];
#pragma unroll
      for (int n = 0; n < 4; ++n) {
        const __bf16* bp = kb + ((gtile * 4 + n) * 32 + lane) * 16;
        bt[n].h[0] = *reinterpret_cast<const v8bf*>(bp);
        bt[n].h[1] = *reinterpret_cast<const v8bf*>(bp + 8);
      }
#pragma unroll
      for (int n = 0; n < 4; ++n) {
        acc[0][gtile * 4 + n] = __builtin_amdgcn_wmma_f32_16x16x32_bf16(
            false, a0, false, bt[n].v, (short)0, acc[0][gtile * 4 + n], false, false);
        acc[1][gtile * 4 + n] = __builtin_amdgcn_wmma_f32_16x16x32_bf16(
            false, a1, false, bt[n].v, (short)0, acc[1][gtile * 4 + n], false, false);
      }
    }
  }

  // ---- epilogue: 4 passes (one per b), sigmoid + LDS stage + b128 stores ----
#pragma unroll 1
  for (int p = 0; p < kNB; ++p) {
    __syncthreads();                    // LDS free for obuf alias / reuse
    if (bLocal == p) {                  // wave-uniform branch
#pragma unroll
      for (int rt = 0; rt < 2; ++rt) {
#pragma unroll
        for (int n = 0; n < kNT; ++n) {
          const int m = n * 16 + l16;
#pragma unroll
          for (int r = 0; r < 8; ++r) {
            const int d = dHalf + rt * 16 + halfSel * 8 + r;  // C layout 7.12.2
            const float v = acc[rt][n][r];
            obuf[m * kD + d] = 1.0f / (1.0f + __expf(-v));
          }
        }
      }
    }
    __syncthreads();
    v4f* o4 = reinterpret_cast<v4f*>(outg + ((size_t)wg * kNB + p) * (kM * kD));
    const v4f* s4 = reinterpret_cast<const v4f*>(obuf);
#pragma unroll
    for (int r = 0; r < 8; ++r) {       // 2048 float4 / 256 threads
      const int q = t + kThreads * r;
      o4[q] = s4[q];
    }
  }
}

extern "C" void kernel_launch(void* const* d_in, const int* in_sizes, int n_in,
                              void* d_out, int out_size, void* d_ws, size_t ws_size,
                              hipStream_t stream) {
  const float* x0 = (const float*)d_in[0];   // [B, 40, 64] f32
  const float* x  = (const float*)d_in[1];   // [B, 40, 64] f32
  const float* k  = (const float*)d_in[2];   // [40, 40, 128] f32
  float* out      = (float*)d_out;           // [B, 128, 64] f32
  __bf16* bpack   = (__bf16*)d_ws;           // 409,600 B packed B table

  prepack_kernel_b<<<(kPackElems + 255) / 256, 256, 0, stream>>>(k, bpack);

  const int B = in_sizes[0] / kXElems;       // 4096
  cin_wmma_bf16_kernel<<<B / kNB, kThreads, kLdsBytes, stream>>>(x0, x, bpack, out);
}
